// TSP_Decoder_38405597561212
// MI455X (gfx1250) — compile-verified, matrix operand order checked by reading, not run on hardware
//
#include <hip/hip_runtime.h>

typedef __attribute__((ext_vector_type(16))) __bf16 v16bf;
typedef __attribute__((ext_vector_type(8)))  __bf16 v8bf;
typedef __attribute__((ext_vector_type(8)))  float  v8f;

#define DEV __device__ __forceinline__

// K index of the low element of VGPR-pair p (0..7) for a given lane-half,
// per CDNA5 ISA 16-bit A-matrix 16x32 layout (CDNA5 ISA 7.12.2).
DEV int kmap(int p, int hf) {
    return (p < 4) ? (p * 2 + hf * 8) : (16 + (p - 4) * 2 + hf * 8);
}

// A fragment (16x32 bf16) from row-major fp32, k contiguous in memory.
DEV v16bf load_afrag_f32(const float* src, int ld, long row, int kbase, int hf) {
    v16bf f;
#pragma unroll
    for (int p = 0; p < 8; ++p) {
        int k = kbase + kmap(p, hf);
        float2 d = *(const float2*)(src + row * (long)ld + k);
        f[2 * p]     = (__bf16)d.x;
        f[2 * p + 1] = (__bf16)d.y;
    }
    return f;
}

// B fragment (32x16) from bf16 stored with k contiguous: src[col*nstride + k].
DEV v16bf load_bfrag_bf16(const __bf16* src, long nstride, int col, int kbase, int hf) {
    v16bf f;
#pragma unroll
    for (int p = 0; p < 8; ++p) {
        int k = kbase + kmap(p, hf);
        const __bf16* q = src + (long)col * nstride + k;
        f[2 * p]     = q[0];
        f[2 * p + 1] = q[1];
    }
    return f;
}

// Same but clamps the pair start so reads never pass kmax (used at the M edge;
// the clamped values are multiplied by e_bias==0 so they never contribute).
DEV v16bf load_bfrag_bf16_cl(const __bf16* src, long nstride, int col, int kbase,
                             int hf, int kmaxpair) {
    v16bf f;
#pragma unroll
    for (int p = 0; p < 8; ++p) {
        int k = kbase + kmap(p, hf);
        if (k > kmaxpair) k = kmaxpair;
        const __bf16* q = src + (long)col * nstride + k;
        f[2 * p]     = q[0];
        f[2 * p + 1] = q[1];
    }
    return f;
}

// Pre-swizzled weight fragment: one 32-byte vector load (2x global_load_b128).
DEV v16bf load_pw(const __bf16* pw, int mat, int ct, int kc, int lane) {
    return *(const v16bf*)(pw + (long)((((mat * 8 + ct) * 4 + kc) * 32 + lane) << 4));
}

DEV v8f wmma_bf16(v16bf a, v16bf b, v8f c) {
    return __builtin_amdgcn_wmma_f32_16x16x32_bf16(false, a, false, b, (short)0, c,
                                                   false, false);
}

// gfx1250 async global->LDS copy (VGLOBAL GLOBAL_LOAD_ASYNC_TO_LDS_B32,
// tracked by ASYNCcnt). vdst = LDS byte address, vaddr = 64-bit global address.
DEV void async_g2l_b32(void* lds_ptr, const void* gptr) {
    unsigned loff = (unsigned)(unsigned long long)lds_ptr;  // low 32b = LDS offset
    asm volatile("global_load_async_to_lds_b32 %0, %1, off"
                 :
                 : "v"(loff), "v"(gptr)
                 : "memory");
}
DEV void wait_async0() { asm volatile("s_wait_asynccnt 0" ::: "memory"); }

// -------------------------------------------------------------------------
// K0: pack the four 128x128 fp32 weights into bf16 WMMA B-fragment order:
// pw[((mat*8 + coltile)*4 + kchunk)*32 + lane][0..15]. mats: 0=Wk 1=Wv 2=Wqf 3=Wql
__global__ void __launch_bounds__(32) pack_w_kernel(const float* __restrict__ W0,
                                                    const float* __restrict__ W1,
                                                    const float* __restrict__ W2,
                                                    const float* __restrict__ W3,
                                                    __bf16* __restrict__ pw) {
    int mat = blockIdx.x, ct = blockIdx.y, kc = blockIdx.z;
    const float* W = (mat == 0) ? W0 : (mat == 1) ? W1 : (mat == 2) ? W2 : W3;
    int lane = threadIdx.x & 31, hf = lane >> 4;
    int col = ct * 16 + (lane & 15);
    __bf16* o = pw + (long)((((mat * 8 + ct) * 4 + kc) * 32 + lane) << 4);
#pragma unroll
    for (int p = 0; p < 8; ++p) {
        int k = kc * 32 + kmap(p, hf);
        o[2 * p]     = (__bf16)W[(long)k * 128 + col];
        o[2 * p + 1] = (__bf16)W[(long)(k + 1) * 128 + col];
    }
}

// -------------------------------------------------------------------------
// K1: ek_t = exp(enc @ Wk), ekv_t = exp(enc @ Wk) * (enc @ Wv), bf16,
// stored transposed as (b, d, m) so the K-dim (m) is contiguous for K3 B-frags.
// grid.x = (64*2000)/16 = 8000 row tiles, 8 waves = 8 d-col tiles.
__global__ void __launch_bounds__(256) kv_kernel(const float* __restrict__ enc,
                                                 const __bf16* __restrict__ pw,
                                                 __bf16* __restrict__ ek_t,
                                                 __bf16* __restrict__ ekv_t) {
    int lane = threadIdx.x & 31, wave = threadIdx.x >> 5, hf = lane >> 4;
    long rowbase = (long)blockIdx.x * 16;
    long arow = rowbase + (lane & 15);
    int col = wave * 16 + (lane & 15);
    v8f ack = {}, acv = {};
#pragma unroll
    for (int kc = 0; kc < 4; ++kc) {
        v16bf a  = load_afrag_f32(enc, 128, arow, kc * 32, hf);
        v16bf bk = load_pw(pw, 0, wave, kc, lane);
        v16bf bv = load_pw(pw, 1, wave, kc, lane);
        ack = wmma_bf16(a, bk, ack);
        acv = wmma_bf16(a, bv, acv);
    }
    int b  = (int)(rowbase / 2000);   // 2000 % 16 == 0: tile never crosses batch
    int m0 = (int)(rowbase - (long)b * 2000);
    v8bf vek, vekv;
#pragma unroll
    for (int i = 0; i < 8; ++i) {
        float ek = __expf(ack[i]);
        vek[i]  = (__bf16)ek;
        vekv[i] = (__bf16)(ek * acv[i]);
    }
    // lane's 8 C rows are consecutive m: one 16B store per matrix.
    long base = ((long)(b * 128 + col)) * 2000 + m0 + 8 * hf;
    *(v8bf*)(ek_t + base)  = vek;
    *(v8bf*)(ekv_t + base) = vekv;
}

// -------------------------------------------------------------------------
// K2: sigq = sigmoid(enc_q1 @ Wq_first + enc_last @ Wq_last), fp32 (b*n, d).
__global__ void __launch_bounds__(256) q_kernel(const float* __restrict__ q1,
                                                const float* __restrict__ qlast,
                                                const __bf16* __restrict__ pw,
                                                float* __restrict__ sigq) {
    int lane = threadIdx.x & 31, wave = threadIdx.x >> 5, hf = lane >> 4;
    long rowbase = (long)blockIdx.x * 16;
    long arow = rowbase + (lane & 15);
    int col = wave * 16 + (lane & 15);
    v8f acc = {};
#pragma unroll
    for (int kc = 0; kc < 4; ++kc) {
        v16bf a1 = load_afrag_f32(q1, 128, arow, kc * 32, hf);
        v16bf b1 = load_pw(pw, 2, wave, kc, lane);
        acc = wmma_bf16(a1, b1, acc);
        v16bf a2 = load_afrag_f32(qlast, 128, arow, kc * 32, hf);
        v16bf b2 = load_pw(pw, 3, wave, kc, lane);
        acc = wmma_bf16(a2, b2, acc);
    }
#pragma unroll
    for (int i = 0; i < 8; ++i) {
        long r = rowbase + i + 8 * hf;
        sigq[r * 128 + col] = 1.0f / (1.0f + __expf(-acc[i]));
    }
}

// -------------------------------------------------------------------------
// K3: AFT module. num/den = e_bias @ (ekv / ek), aafm = sigq * num/den (bf16).
// grid (64 batches, 7 n-tiles); 8 waves = 8 d-tiles.
// cur_dist tile staged via gfx1250 async global->LDS, e_bias tile in padded LDS.
__global__ void __launch_bounds__(256) aafm_kernel(const float* __restrict__ cur_dist,
                                                   const float* __restrict__ ninf,
                                                   const __bf16* __restrict__ ek_t,
                                                   const __bf16* __restrict__ ekv_t,
                                                   const float* __restrict__ sigq,
                                                   const float* __restrict__ alpha1,
                                                   const float* __restrict__ log_scale,
                                                   __bf16* __restrict__ aafm) {
    __shared__ float  dstage[16 * 32];   // raw cur_dist tile (async landing zone)
    __shared__ __bf16 ebias[16 * 34];    // pitch 34 (17 dwords) -> conflict-free
    int b = blockIdx.x, ntile = blockIdx.y;
    int lane = threadIdx.x & 31, wave = threadIdx.x >> 5, hf = lane >> 4;
    float c1 = -alpha1[0] * log_scale[0];
    const __bf16* ekb  = ek_t  + (long)b * 128 * 2000;
    const __bf16* ekvb = ekv_t + (long)b * 128 * 2000;
    int dcol = wave * 16 + (lane & 15);
    v8f an = {}, ad = {};
    for (int m0 = 0; m0 < 2000; m0 += 32) {
        __syncthreads();  // protect ebias/dstage reuse across iterations
        // async-copy the raw 16x32 cur_dist tile into LDS (2 elements/thread)
        for (int t = threadIdx.x; t < 512; t += 256) {
            int r = t >> 5, c = t & 31;
            int n = ntile * 16 + r; n = (n > 99) ? 99 : n;
            int m = m0 + c;         int mc = (m > 1999) ? 1999 : m;
            async_g2l_b32(&dstage[t], cur_dist + ((long)b * 100 + n) * 2000 + mc);
        }
        wait_async0();
        // exp + bf16 convert out of LDS (each thread touches its own elements)
        for (int t = threadIdx.x; t < 512; t += 256) {
            int r = t >> 5, c = t & 31;
            int n = ntile * 16 + r; n = (n > 99) ? 99 : n;
            int m = m0 + c;         int mc = (m > 1999) ? 1999 : m;
            float nv = ninf[((long)b * 100 + n) * 2000 + mc];
            float eb = (m < 2000) ? __expf(c1 * dstage[t] + nv) : 0.0f;
            ebias[r * 34 + c] = (__bf16)eb;
        }
        __syncthreads();
        v16bf a;
        int r = lane & 15;
#pragma unroll
        for (int p = 0; p < 8; ++p) {
            int k = kmap(p, hf);
            a[2 * p]     = ebias[r * 34 + k];
            a[2 * p + 1] = ebias[r * 34 + k + 1];
        }
        v16bf bn = load_bfrag_bf16_cl(ekvb, 2000, dcol, m0, hf, 1998);
        v16bf bd = load_bfrag_bf16_cl(ekb,  2000, dcol, m0, hf, 1998);
        an = wmma_bf16(a, bn, an);
        ad = wmma_bf16(a, bd, ad);
    }
#pragma unroll
    for (int i = 0; i < 8; ++i) {
        int n = ntile * 16 + i + 8 * hf;
        if (n < 100) {
            long off = ((long)b * 100 + n) * 128 + dcol;
            float w = an[i] / ad[i];
            aafm[off] = (__bf16)(sigq[off] * w);
        }
    }
}

// -------------------------------------------------------------------------
// K4: score^T tiles (m rows x n cols) so the softmax axis = C rows; writes
// 10*tanh(score/sqrt(128) - a2*ls*dist) + ninf into out (f32).
__global__ void __launch_bounds__(256) score_kernel(const float* __restrict__ enc,
                                                    const __bf16* __restrict__ aafm,
                                                    const float* __restrict__ cur_dist,
                                                    const float* __restrict__ ninf,
                                                    const float* __restrict__ alpha2,
                                                    const float* __restrict__ log_scale,
                                                    float* __restrict__ out) {
    int lane = threadIdx.x & 31, wave = threadIdx.x >> 5, hf = lane >> 4;
    int b = blockIdx.x, ntile = blockIdx.y;
    int mt = blockIdx.z * 8 + wave;
    if (mt >= 125) return;
    const float*  encb = enc  + (long)b * 2000 * 128;
    const __bf16* ab   = aafm + (long)b * 100 * 128;
    long mrow = (long)mt * 16 + (lane & 15);
    int ncol = ntile * 16 + (lane & 15); ncol = (ncol > 99) ? 99 : ncol;
    v8f acc = {};
#pragma unroll
    for (int kk = 0; kk < 128; kk += 32) {
        if (kk < 96) __builtin_prefetch(encb + mrow * 128 + kk + 32, 0, 0);
        v16bf a  = load_afrag_f32(encb, 128, mrow, kk, hf);
        v16bf bf = load_bfrag_bf16(ab, 128, ncol, kk, hf);
        acc = wmma_bf16(a, bf, acc);
    }
    float c2 = -alpha2[0] * log_scale[0];
#pragma unroll
    for (int i = 0; i < 8; ++i) {
        int m = mt * 16 + i + 8 * hf;
        int n = ntile * 16 + (lane & 15);
        if (n < 100) {
            long off = ((long)b * 100 + n) * 2000 + m;
            float s = acc[i] * 0.08838834764831845f + c2 * cur_dist[off];
            out[off] = 10.0f * tanhf(s) + ninf[off];
        }
    }
}

// -------------------------------------------------------------------------
// K5: row softmax over M=2000, in place on out. One block per (b,n) row.
__global__ void __launch_bounds__(256) softmax_kernel(float* __restrict__ out) {
    long base = (long)blockIdx.x * 2000;
    int tid = threadIdx.x, lane = tid & 31, wave = tid >> 5;
    __shared__ float redm[8];
    __shared__ float reds[8];
    float v[8];
    float mx = -1e30f;
#pragma unroll
    for (int j = 0; j < 8; ++j) {
        int i = tid + j * 256;
        if (i < 2000) { v[j] = out[base + i]; mx = fmaxf(mx, v[j]); }
        else v[j] = -1e30f;
    }
    for (int o = 16; o > 0; o >>= 1) mx = fmaxf(mx, __shfl_xor(mx, o, 32));
    if (lane == 0) redm[wave] = mx;
    __syncthreads();
    if (tid == 0) {
        float m = redm[0];
        for (int w = 1; w < 8; ++w) m = fmaxf(m, redm[w]);
        redm[0] = m;
    }
    __syncthreads();
    mx = redm[0];
    float sum = 0.0f;
#pragma unroll
    for (int j = 0; j < 8; ++j) {
        int i = tid + j * 256;
        if (i < 2000) { v[j] = __expf(v[j] - mx); sum += v[j]; }
    }
    for (int o = 16; o > 0; o >>= 1) sum += __shfl_xor(sum, o, 32);
    if (lane == 0) reds[wave] = sum;
    __syncthreads();
    if (tid == 0) {
        float s = 0.0f;
        for (int w = 0; w < 8; ++w) s += reds[w];
        reds[0] = s;
    }
    __syncthreads();
    float inv = 1.0f / reds[0];
#pragma unroll
    for (int j = 0; j < 8; ++j) {
        int i = tid + j * 256;
        if (i < 2000) out[base + i] = v[j] * inv;
    }
}

extern "C" void kernel_launch(void* const* d_in, const int* in_sizes, int n_in,
                              void* d_out, int out_size, void* d_ws, size_t ws_size,
                              hipStream_t stream) {
    const float* enc      = (const float*)d_in[0];   // (64,2000,128)
    const float* q1       = (const float*)d_in[1];   // (64,100,128)
    const float* qlast    = (const float*)d_in[2];   // (64,100,128)
    const float* cur_dist = (const float*)d_in[3];   // (64,100,2000)
    const float* lscale   = (const float*)d_in[4];   // (1,)
    const float* ninf     = (const float*)d_in[5];   // (64,100,2000)
    const float* Wqf      = (const float*)d_in[6];   // (128,128)
    const float* Wql      = (const float*)d_in[7];
    const float* Wk       = (const float*)d_in[8];
    const float* Wv       = (const float*)d_in[9];
    const float* a1       = (const float*)d_in[10];
    const float* a2       = (const float*)d_in[11];
    float* out = (float*)d_out;                      // (64,100,2000)

    char* ws = (char*)d_ws;
    __bf16* ek_t  = (__bf16*)ws;                        // 32.768 MB
    __bf16* ekv_t = (__bf16*)(ws + 32768000);           // 32.768 MB
    float*  sigq  = (float*)(ws + 65536000);            // 3.2768 MB
    __bf16* aafm  = (__bf16*)(ws + 65536000 + 3276800); // 1.6384 MB
    __bf16* pw    = (__bf16*)(ws + 70451200);           // 128 KB packed weights

    pack_w_kernel<<<dim3(4, 8, 4), 32, 0, stream>>>(Wk, Wv, Wqf, Wql, pw);
    kv_kernel<<<8000, 256, 0, stream>>>(enc, pw, ek_t, ekv_t);
    q_kernel<<<400, 256, 0, stream>>>(q1, qlast, pw, sigq);
    aafm_kernel<<<dim3(64, 7), 256, 0, stream>>>(cur_dist, ninf, ek_t, ekv_t, sigq,
                                                 a1, lscale, aafm);
    score_kernel<<<dim3(64, 7, 16), 256, 0, stream>>>(enc, aafm, cur_dist, ninf,
                                                      a2, lscale, out);
    softmax_kernel<<<6400, 256, 0, stream>>>(out);
}
